// CrossAttention_79448305041860
// MI455X (gfx1250) — matrix operand order repaired and checked by reading, not compiled, not run on hardware
//
#include <hip/hip_runtime.h>

typedef __attribute__((ext_vector_type(16))) __bf16 v16bf;
typedef __attribute__((ext_vector_type(8)))  float  v8f;

#define DEVFN __device__ __forceinline__

struct FragBF { union { v16bf v; uint4 u[2]; }; };

DEVFN unsigned short f2bf(float x) {
  unsigned int u = __builtin_bit_cast(unsigned int, x);
  unsigned int r = u + 0x7FFFu + ((u >> 16) & 1u);   // round-to-nearest-even
  return (unsigned short)(r >> 16);
}

// A-matrix 16x32 bf16 fragment (ISA 7.12.2): lanes 0-15 hold K {0..7,16..23},
// lanes 16-31 hold K {8..15,24..31}; row M = lane%16. Source row-major [M][K].
DEVFN v16bf load_a_frag(const unsigned short* p, int ld) {
  const int lane = threadIdx.x & 31;
  const unsigned short* q = p + (lane & 15) * ld + ((lane >> 4) << 3);
  FragBF f;
  f.u[0] = *(const uint4*)(q);
  f.u[1] = *(const uint4*)(q + 16);
  return f.v;
}

// B-matrix 32x16 bf16 fragment: col N = lane%16; lanes 0-15 K {0..7,16..23},
// lanes 16-31 K {8..15,24..31}. Source layout: element (k,n) at p[n*ldn + k].
DEVFN v16bf load_b_frag(const unsigned short* p, int ldn) {
  const int lane = threadIdx.x & 31;
  const unsigned short* q = p + (lane & 15) * ldn + ((lane >> 4) << 3);
  FragBF f;
  f.u[0] = *(const uint4*)(q);
  f.u[1] = *(const uint4*)(q + 16);
  return f.v;
}

DEVFN v8f wmma_bf16(v16bf a, v16bf b, v8f c) {
  return __builtin_amdgcn_wmma_f32_16x16x32_bf16(
      false, a, false, b, (short)0, c, false, false);
}

// gfx1250 async global->LDS copy (16B per lane), tracked by ASYNCcnt.
// LDS offset = low 32 bits of the flat shared-pointer (aperture in [63:32]).
DEVFN void async_copy16(unsigned short* lds_dst, const unsigned short* gsrc) {
#if defined(__HIP_DEVICE_COMPILE__)
  unsigned int loff = (unsigned int)(size_t)lds_dst;
  unsigned long long ga = (unsigned long long)(size_t)gsrc;
  asm volatile("global_load_async_to_lds_b128 %0, %1, off"
               :: "v"(loff), "v"(ga) : "memory");
#else
  *(uint4*)lds_dst = *(const uint4*)gsrc;
#endif
}

DEVFN void wait_async0() {
#if defined(__HIP_DEVICE_COMPILE__)
  asm volatile("s_wait_asynccnt 0x0" ::: "memory");
#endif
}

// ---------------------------------------------------------------- weights->bf16
__global__ __launch_bounds__(256) void convert_bf16_kernel(
    const float* __restrict__ src, unsigned short* __restrict__ dst, int n) {
  int i = blockIdx.x * 256 + threadIdx.x;
  if (i < n) dst[i] = f2bf(src[i]);
}

// ---------------------------------------------------------------- projections
// Y[b,n,e] = sum_c X[b,c,n] * W[e,c] + bias[e]; X channel-major [C][4096].
// transOut=0 -> Y[b][n][e]; transOut=1 -> Y[b][e][n] (for V^T).
template <int C>
__global__ __launch_bounds__(256) void proj_kernel(
    const float* __restrict__ X, const unsigned short* __restrict__ Wbf,
    const float* __restrict__ bias, unsigned short* __restrict__ Y,
    int transOut) {
  constexpr int N = 4096, E = 256;
  constexpr int LDA = C + 8;                 // pad: 528B/1040B row stride (bank step != 0 mod 64)
  __shared__ __align__(16) unsigned short sA[16 * LDA];   // token tile [n][c]
  const int b  = blockIdx.y;
  const int n0 = blockIdx.x * 16;
  const float* Xb = X + (size_t)b * C * N;
  for (int idx = threadIdx.x; idx < 16 * C; idx += 256) {
    int n = idx & 15;
    int c = idx >> 4;
    sA[n * LDA + c] = f2bf(Xb[(size_t)c * N + n0 + n]);
  }
  __syncthreads();

  const int wave = threadIdx.x >> 5;
  const int lane = threadIdx.x & 31;
  const int e0 = wave * 32;                  // 8 waves * 32 = 256 = E
  v8f acc0 = {}, acc1 = {};
  for (int c = 0; c < C; c += 32) {
    v16bf a  = load_a_frag(sA + c, LDA);
    v16bf b0 = load_b_frag(Wbf + (size_t)e0 * C + c, C);
    v16bf b1 = load_b_frag(Wbf + (size_t)(e0 + 16) * C + c, C);
    acc0 = wmma_bf16(a, b0, acc0);
    acc1 = wmma_bf16(a, b1, acc1);
  }
  const int ec0 = e0 + (lane & 15), ec1 = e0 + 16 + (lane & 15);
  const float bi0 = bias[ec0], bi1 = bias[ec1];
#pragma unroll
  for (int r = 0; r < 8; ++r) {
    int mrow = r + ((lane >> 4) << 3);       // C/D layout: M = r + 8*(lane>=16)
    float y0 = acc0[r] + bi0;
    float y1 = acc1[r] + bi1;
    if (!transOut) {
      Y[((size_t)b * N + n0 + mrow) * E + ec0] = f2bf(y0);
      Y[((size_t)b * N + n0 + mrow) * E + ec1] = f2bf(y1);
    } else {
      Y[((size_t)b * E + ec0) * N + n0 + mrow] = f2bf(y0);
      Y[((size_t)b * E + ec1) * N + n0 + mrow] = f2bf(y1);
    }
  }
}

// ---------------------------------------------------------------- flash attention
// 256 thr = 8 waves; 4 query groups of 16 rows x 2 waves (128-wide E halves).
// Key/V tiles (32 keys) double-buffered in LDS, prefetched with
// global_load_async_to_lds_b128 and overlapped with the WMMA chain.
constexpr int AN = 4096, AE = 256;
constexpr int LDK = AE + 8;   // 528B rows: lane rows land in distinct banks
constexpr int LDV = 40;       // 80B rows:  bank step 20, 16 rows start in distinct banks

__global__ __launch_bounds__(256) void attn_kernel(
    const unsigned short* __restrict__ Qbf,   // [B][N][E]
    const unsigned short* __restrict__ Kbf,   // [B][N][E]
    const unsigned short* __restrict__ Vt,    // [B][E][N]  (transposed)
    unsigned short* __restrict__ comb,        // [B][N][512]
    int eofs) {
  __shared__ __align__(16) unsigned short sK[2][32 * LDK];
  __shared__ __align__(16) unsigned short sV[2][AE * LDV];
  __shared__ __align__(16) unsigned short sP[4][16 * LDV];

  const int b     = blockIdx.y;
  const int wave  = threadIdx.x >> 5;
  const int lane  = threadIdx.x & 31;
  const int qg    = wave >> 1;
  const int ehalf = (wave & 1) * 128;
  const int q0    = blockIdx.x * 64 + qg * 16;
  unsigned short* sPg = sP[qg];

  // async stage of one 32-key tile: K rows [key][e], V rows [e][key]
  auto stage = [&](int buf, int kt) {
    int t = threadIdx.x;
    {
      int key = t >> 3;
      int ee  = (t & 7) * 32;
      const unsigned short* src = Kbf + ((size_t)b * AN + kt + key) * AE + ee;
      unsigned short* dst = &sK[buf][key * LDK + ee];
#pragma unroll
      for (int i = 0; i < 4; ++i) async_copy16(dst + 8 * i, src + 8 * i);
    }
    {
      const unsigned short* src = Vt + ((size_t)b * AE + t) * AN + kt;
      unsigned short* dst = &sV[buf][t * LDV];
#pragma unroll
      for (int i = 0; i < 4; ++i) async_copy16(dst + 8 * i, src + 8 * i);
    }
  };

  v16bf qf[8];
  const unsigned short* qbase = Qbf + ((size_t)b * AN + q0) * AE;
#pragma unroll
  for (int i = 0; i < 8; ++i) qf[i] = load_a_frag(qbase + 32 * i, AE);

  v8f o[8];
#pragma unroll
  for (int j = 0; j < 8; ++j) o[j] = (v8f){};
  float mrun[8], lrun[8];
#pragma unroll
  for (int r = 0; r < 8; ++r) { mrun[r] = -3.0e30f; lrun[r] = 0.f; }
  const float scale = 0.0625f;               // 1/sqrt(256)

  stage(0, 0);

  for (int kt = 0; kt < AN; kt += 32) {
    const int cur = (kt >> 5) & 1;
    wait_async0();                            // this wave's tile loads landed
    __syncthreads();                          // everyone's tile loads landed
    if (kt + 32 < AN) stage(cur ^ 1, kt + 32); // prefetch under compute
    const unsigned short* sKc = sK[cur];
    const unsigned short* sVc = sV[cur];

    // S = Q K^T over full E (two 16-key fragments), B-frags pipelined 1 ahead
    v8f S0 = {}, S1 = {};
    v16bf bk0 = load_b_frag(sKc, LDK);
    v16bf bk1 = load_b_frag(sKc + 16 * LDK, LDK);
#pragma unroll
    for (int i = 0; i < 8; ++i) {
      v16bf nb0, nb1;
      if (i < 7) {
        nb0 = load_b_frag(sKc + 32 * (i + 1), LDK);
        nb1 = load_b_frag(sKc + 16 * LDK + 32 * (i + 1), LDK);
      }
      S0 = wmma_bf16(qf[i], bk0, S0);
      S1 = wmma_bf16(qf[i], bk1, S1);
      bk0 = nb0; bk1 = nb1;
    }

    // online softmax; row m = r + 8*(lane>=16), 16-lane half-wave reductions
#pragma unroll
    for (int r = 0; r < 8; ++r) {
      float s0 = S0[r] * scale;
      float s1 = S1[r] * scale;
      float mx = fmaxf(s0, s1);
#pragma unroll
      for (int m = 1; m < 16; m <<= 1) mx = fmaxf(mx, __shfl_xor(mx, m, 32));
      float mnew  = fmaxf(mrun[r], mx);
      float alpha = __expf(mrun[r] - mnew);
      float p0 = __expf(s0 - mnew);
      float p1 = __expf(s1 - mnew);
      float ls = p0 + p1;
#pragma unroll
      for (int m = 1; m < 16; m <<= 1) ls += __shfl_xor(ls, m, 32);
      lrun[r] = lrun[r] * alpha + ls;
      mrun[r] = mnew;
#pragma unroll
      for (int j = 0; j < 8; ++j) o[j][r] *= alpha;
      int mrow = r + ((lane >> 4) << 3);
      sPg[mrow * LDV + (lane & 15)]      = f2bf(p0);
      sPg[mrow * LDV + 16 + (lane & 15)] = f2bf(p1);
    }

    // O += P V  (this wave's 128-wide E half), B-frags pipelined 1 ahead
    v16bf pa = load_a_frag(sPg, LDV);
    v16bf bv = load_b_frag(sVc + ehalf * LDV, LDV);
#pragma unroll
    for (int j = 0; j < 8; ++j) {
      v16bf nbv;
      if (j < 7) nbv = load_b_frag(sVc + (ehalf + 16 * (j + 1)) * LDV, LDV);
      o[j] = wmma_bf16(pa, bv, o[j]);
      bv = nbv;
    }
  }

  float inv[8];
#pragma unroll
  for (int r = 0; r < 8; ++r) inv[r] = 1.0f / lrun[r];
#pragma unroll
  for (int j = 0; j < 8; ++j) {
#pragma unroll
    for (int r = 0; r < 8; ++r) {
      int mrow = r + ((lane >> 4) << 3);
      int e = ehalf + 16 * j + (lane & 15);
      comb[((size_t)b * AN + q0 + mrow) * 512 + eofs + e] = f2bf(o[j][r] * inv[r]);
    }
  }
}

// ---------------------------------------------------------------- outproj + LN
__global__ __launch_bounds__(256) void outproj_ln_kernel(
    const unsigned short* __restrict__ comb,  // [B][N][512] bf16
    const unsigned short* __restrict__ wobf,  // [E][512] bf16
    const float* __restrict__ bo,
    const float* __restrict__ lnw, const float* __restrict__ lnb,
    float* __restrict__ out) {                // [B][E][N]
  constexpr int N = 4096, E = 256, F = 512;
  constexpr int LDA = F + 8;                  // 1040B rows
  constexpr int LDO = E + 8;
  __shared__ __align__(16) unsigned short sA[16 * LDA];
  __shared__ float sO[16 * LDO];
  const int b  = blockIdx.y;
  const int n0 = blockIdx.x * 16;
  {
    int row = threadIdx.x >> 4;
    int f0  = (threadIdx.x & 15) * 32;
    const unsigned short* src = comb + ((size_t)b * N + n0 + row) * F + f0;
    unsigned short* dst = sA + row * LDA + f0;
#pragma unroll
    for (int i = 0; i < 4; ++i) async_copy16(dst + 8 * i, src + 8 * i);
  }
  wait_async0();
  __syncthreads();

  const int wave = threadIdx.x >> 5;
  const int lane = threadIdx.x & 31;
  const int e0 = wave * 32;
  v8f acc0 = {}, acc1 = {};
  for (int f = 0; f < F; f += 32) {
    v16bf a  = load_a_frag(sA + f, LDA);
    v16bf b0 = load_b_frag(wobf + (size_t)e0 * F + f, F);
    v16bf b1 = load_b_frag(wobf + (size_t)(e0 + 16) * F + f, F);
    acc0 = wmma_bf16(a, b0, acc0);
    acc1 = wmma_bf16(a, b1, acc1);
  }
  const int ec0 = e0 + (lane & 15), ec1 = e0 + 16 + (lane & 15);
  const float bi0 = bo[ec0], bi1 = bo[ec1];
#pragma unroll
  for (int r = 0; r < 8; ++r) {
    int mrow = r + ((lane >> 4) << 3);
    sO[mrow * LDO + ec0] = acc0[r] + bi0;
    sO[mrow * LDO + ec1] = acc1[r] + bi1;
  }
  __syncthreads();

  // LayerNorm over E: 16 threads per row (one row per 16-lane half)
  const int row = threadIdx.x >> 4;
  const int sub = threadIdx.x & 15;
  float sum = 0.f;
#pragma unroll
  for (int i = 0; i < 16; ++i) sum += sO[row * LDO + sub + 16 * i];
#pragma unroll
  for (int m = 1; m < 16; m <<= 1) sum += __shfl_xor(sum, m, 32);
  float mu = sum * (1.0f / E);
  float vs = 0.f;
#pragma unroll
  for (int i = 0; i < 16; ++i) {
    float d = sO[row * LDO + sub + 16 * i] - mu;
    vs += d * d;
  }
#pragma unroll
  for (int m = 1; m < 16; m <<= 1) vs += __shfl_xor(vs, m, 32);
  float rs = rsqrtf(vs * (1.0f / E) + 1e-5f);
#pragma unroll
  for (int i = 0; i < 16; ++i) {
    int e = sub + 16 * i;
    float v = (sO[row * LDO + e] - mu) * rs * lnw[e] + lnb[e];
    out[(size_t)b * E * N + (size_t)e * N + n0 + row] = v;
  }
}

// ---------------------------------------------------------------- launch
extern "C" void kernel_launch(void* const* d_in, const int* in_sizes, int n_in,
                              void* d_out, int out_size, void* d_ws, size_t ws_size,
                              hipStream_t stream) {
  (void)in_sizes; (void)n_in; (void)out_size; (void)ws_size;
  const float* q1  = (const float*)d_in[0];
  const float* q2  = (const float*)d_in[1];
  const float* kv  = (const float*)d_in[2];
  const float* wq1 = (const float*)d_in[3];
  const float* bq1 = (const float*)d_in[4];
  const float* wq2 = (const float*)d_in[5];
  const float* bq2 = (const float*)d_in[6];
  const float* wk  = (const float*)d_in[7];
  const float* bk  = (const float*)d_in[8];
  const float* wv  = (const float*)d_in[9];
  const float* bv  = (const float*)d_in[10];
  const float* wo  = (const float*)d_in[11];
  const float* bo  = (const float*)d_in[12];
  const float* lnw = (const float*)d_in[13];
  const float* lnb = (const float*)d_in[14];
  float* out = (float*)d_out;

  constexpr int B = 4, N = 4096, E = 256;
  char* ws = (char*)d_ws;
  size_t off = 0;
  auto carve = [&](size_t bytes) -> char* {
    char* p = ws + off;
    off += (bytes + 255) & ~(size_t)255;
    return p;
  };
  unsigned short* wq1b = (unsigned short*)carve((size_t)E * 256 * 2);
  unsigned short* wq2b = (unsigned short*)carve((size_t)E * 256 * 2);
  unsigned short* wkb  = (unsigned short*)carve((size_t)E * 512 * 2);
  unsigned short* wvb  = (unsigned short*)carve((size_t)E * 512 * 2);
  unsigned short* wob  = (unsigned short*)carve((size_t)E * 512 * 2);
  unsigned short* Q1b  = (unsigned short*)carve((size_t)B * N * E * 2);
  unsigned short* Q2b  = (unsigned short*)carve((size_t)B * N * E * 2);
  unsigned short* Kb   = (unsigned short*)carve((size_t)B * N * E * 2);
  unsigned short* Vtb  = (unsigned short*)carve((size_t)B * E * N * 2);
  unsigned short* comb = (unsigned short*)carve((size_t)B * N * 512 * 2);

  auto cvt = [&](const float* s, unsigned short* d, int n) {
    convert_bf16_kernel<<<(n + 255) / 256, 256, 0, stream>>>(s, d, n);
  };
  cvt(wq1, wq1b, E * 256);
  cvt(wq2, wq2b, E * 256);
  cvt(wk,  wkb,  E * 512);
  cvt(wv,  wvb,  E * 512);
  cvt(wo,  wob,  E * 512);

  dim3 pg(N / 16, B);
  proj_kernel<256><<<pg, 256, 0, stream>>>(q1, wq1b, bq1, Q1b, 0);
  proj_kernel<256><<<pg, 256, 0, stream>>>(q2, wq2b, bq2, Q2b, 0);
  proj_kernel<512><<<pg, 256, 0, stream>>>(kv, wkb,  bk,  Kb,  0);
  proj_kernel<512><<<pg, 256, 0, stream>>>(kv, wvb,  bv,  Vtb, 1);  // transposed

  dim3 ag(N / 64, B);
  attn_kernel<<<ag, 256, 0, stream>>>(Q1b, Kb, Vtb, comb, 0);
  attn_kernel<<<ag, 256, 0, stream>>>(Q2b, Kb, Vtb, comb, 256);

  dim3 og(N / 16, B);
  outproj_ln_kernel<<<og, 256, 0, stream>>>(comb, wob, bo, lnw, lnb, out);
}
